// ModelEncodeDecodeSimple_42056319763003
// MI455X (gfx1250) — compile-verified, matrix operand order checked by reading, not run on hardware
//
// Spherical CNN (S2 conv + 3x SO3 conv + BN/PReLU + quadrature) for MI455X (gfx1250).
//
// Strategy: all FFT/iFFT stages are executed as dense complex DFT matmuls on the
// wave32 WMMA unit (v_wmma_f32_16x16x32_f16, f16 twiddles / f32 accumulate).
// Every intermediate tensor (<=138 MB planar complex f32) fits in the 192 MB L2,
// so the net is bandwidth bound and the extra DFT FLOPs are free on the matrix
// pipes. The GEMM stages its streaming A operand into LDS with
// GLOBAL_LOAD_ASYNC_TO_LDS_B128 (ASYNCcnt-tracked DMA) on contiguous-k passes,
// and with row-coalesced scalar loads on the strided "column DFT" passes.
// Wigner-d banks, quadrature weights and kernel spectra are generated on device
// (the reference builds them host-side with numpy; they are not inputs).
// Workspace requirement: ~470 MB.

#include <hip/hip_runtime.h>
#include <cstdint>

typedef __attribute__((ext_vector_type(16))) _Float16 v16h;
typedef __attribute__((ext_vector_type(8)))  float    v8f;
typedef long long i64;

#define PI_D 3.14159265358979323846
#define BIGI 0x3fffffff

__host__ __device__ static inline i64 cum_sq(int l)  { return (i64)l * (2*l - 1) * (2*l + 1) / 3; } // sum_{k<l}(2k+1)^2
__host__ __device__ static inline i64 cum_lin(int l) { return (i64)l * l; }                          // sum_{k<l}(2k+1)

__device__ static inline i64 gidx() { return (i64)blockIdx.x * blockDim.x + threadIdx.x; }

// ---------------------------------------------------------------------------
// Table generation: log-factorials, quadrature weights, DFT matrices, Wigner-d
// ---------------------------------------------------------------------------
__global__ void sphcnn_lfact(double* lf) {
  if (blockIdx.x == 0 && threadIdx.x == 0) {
    lf[0] = 0.0;
    for (int i = 1; i < 128; ++i) lf[i] = lf[i - 1] + log((double)i);
  }
}

__device__ static double wig_d(int l, int mp, int m, double beta, const double* lf) {
  // d^l_{mp,m}(beta), explicit factorial formula (l <= 29 -> double is fine)
  double cb = cos(0.5 * beta), sb = sin(0.5 * beta);
  int smin = (m - mp) > 0 ? (m - mp) : 0;
  int smax = (l + m) < (l - mp) ? (l + m) : (l - mp);
  if (smax < smin) return 0.0;
  double pref = 0.5 * (lf[l + mp] + lf[l - mp] + lf[l + m] + lf[l - m]);
  double acc = 0.0;
  for (int s = smin; s <= smax; ++s) {
    double lg = pref - lf[l + m - s] - lf[s] - lf[mp - m + s] - lf[l - mp - s];
    double t = exp(lg) * pow(cb, (double)(2 * l + m - mp - 2 * s)) * pow(sb, (double)(mp - m + 2 * s));
    acc += ((mp - m + s) & 1) ? -t : t;
  }
  return acc;
}

__global__ void sphcnn_quadw(float* w, int b) {
  i64 j = gidx();
  if (j >= 2 * b) return;
  double s = 0.0;
  for (int k = 0; k < b; ++k)
    s += sin((2.0 * j + 1.0) * (2.0 * k + 1.0) * PI_D / (4.0 * b)) / (2.0 * k + 1.0);
  w[j] = (float)(2.0 / b * sin(PI_D * (2.0 * j + 1.0) / (4.0 * b)) * s);
}

__global__ void sphcnn_dftmat(float* re, float* im, int N, int sign) {
  i64 idx = gidx();
  if (idx >= (i64)N * N) return;
  int a = (int)(idx / N), m = (int)(idx % N);
  double th = 2.0 * PI_D * (double)a * (double)m / (double)N;
  re[idx] = (float)cos(th);
  im[idx] = (float)(sign * sin(th));
}

// SO3 bank: layout [l][j][m][n]; analysis (*w[j]) or synthesis (*(2l+1))
__global__ void sphcnn_so3_bank(float* out, int twob, int nl, const float* w, int synth,
                                const double* lf, i64 total) {
  i64 idx = gidx();
  if (idx >= total) return;
  int l = 0;
  while ((i64)twob * cum_sq(l + 1) <= idx) ++l;
  i64 rem = idx - (i64)twob * cum_sq(l);
  int d = 2 * l + 1;
  int j  = (int)(rem / ((i64)d * d));
  int r2 = (int)(rem % ((i64)d * d));
  int mi = r2 / d, ni = r2 % d;
  double beta = PI_D * (2.0 * j + 1.0) / (2.0 * twob);
  double v = wig_d(l, mi - l, ni - l, beta, lf);
  out[idx] = (float)(v * (synth ? (double)(2 * l + 1) : (double)w[j]));
}

// S2 analysis bank: layout [l][j][m]  (n = 0 column), * w[j]
__global__ void sphcnn_s2_bank(float* out, int twob, int nl, const float* w,
                               const double* lf, i64 total) {
  i64 idx = gidx();
  if (idx >= total) return;
  int l = 0;
  while ((i64)twob * cum_lin(l + 1) <= idx) ++l;
  i64 rem = idx - (i64)twob * cum_lin(l);
  int d = 2 * l + 1;
  int j = (int)(rem / d), mi = (int)(rem % d);
  double beta = PI_D * (2.0 * j + 1.0) / (2.0 * twob);
  out[idx] = (float)(wig_d(l, mi - l, 0, beta, lf) * (double)w[j]);
}

// ---------------------------------------------------------------------------
// Kernel spectra Fy (tiny; grids have a single beta, 6 alphas [x 6 gammas])
// ---------------------------------------------------------------------------
__global__ void sphcnn_spec_s2(float* Fre, float* Fim, const float* kk, int nl, int Ci, int Co,
                               double beta, int nA, float scale, const double* lf, i64 total) {
  i64 idx = gidx();
  if (idx >= total) return;
  i64 row = idx / (Ci * Co);
  int io = (int)(idx % (Ci * Co));
  int i = io / Co, o = io % Co;
  int l = 0;
  while (cum_lin(l + 1) <= row) ++l;
  int m = (int)(row - cum_lin(l)) - l;
  double d = wig_d(l, m, 0, beta, lf);
  double sre = 0.0, sim = 0.0;
  for (int g = 0; g < nA; ++g) {
    double alpha = 2.0 * PI_D * g / nA;
    double kv = (double)kk[((i64)i * Co + o) * nA + g];
    sre += kv * d * cos(alpha * m);   // conj(Y) = d * e^{+i alpha m}
    sim += kv * d * sin(alpha * m);
  }
  Fre[idx] = (float)(scale * sre);
  Fim[idx] = (float)(scale * sim);
}

__global__ void sphcnn_spec_so3(float* Fre, float* Fim, const float* kk, int nl, int Ci, int Co,
                                double beta, int nA, int nG, float scale, const double* lf, i64 total) {
  i64 idx = gidx();
  if (idx >= total) return;
  i64 row = idx / (Ci * Co);
  int io = (int)(idx % (Ci * Co));
  int i = io / Co, o = io % Co;
  int l = 0;
  while (cum_sq(l + 1) <= row) ++l;
  i64 rem = row - cum_sq(l);
  int d = 2 * l + 1;
  int m = (int)(rem / d) - l, n = (int)(rem % d) - l;
  double wd = wig_d(l, m, n, beta, lf);
  int G = nA * nG;
  double sre = 0.0, sim = 0.0;
  for (int g = 0; g < G; ++g) {
    double alpha = 2.0 * PI_D * (g / nG) / nA;
    double gamma = 2.0 * PI_D * (g % nG) / nG;
    double ph = alpha * m + gamma * n;            // conj(D) = d * e^{+i(alpha m + gamma n)}
    double kv = (double)kk[((i64)i * Co + o) * G + g];
    sre += kv * wd * cos(ph);
    sim += kv * wd * sin(ph);
  }
  Fre[idx] = (float)(scale * sre);
  Fim[idx] = (float)(scale * sim);
}

// ---------------------------------------------------------------------------
// WMMA complex GEMM:  C[M,N] = A[M,K] * B[K,N]   (B = dense DFT matrix, K x N)
// 4 waves / block, 64x16 C macro-tile. A tile staged into LDS:
//  - contiguous-k pass : GLOBAL_LOAD_ASYNC_TO_LDS_B128 (ASYNCcnt DMA)
//  - strided-k pass    : row-coalesced scalar loads (row stride aSI == 1)
// A/C addressed via (row/inner)*sO + (row%inner)*sI + k*sK  -> handles both the
// contiguous "row DFT" pass and the strided "column DFT" pass without transposes.
// ---------------------------------------------------------------------------
__global__ __launch_bounds__(128)
void sphcnn_gemm_cplx_wmma(const float* __restrict__ Are, const float* __restrict__ Aim,
                           const float* __restrict__ Bre, const float* __restrict__ Bim,
                           float* __restrict__ Cre, float* __restrict__ Cim,
                           int M, int N, int K,
                           int aInner, i64 aSI, i64 aSO, i64 aSK,
                           int cInner, i64 cSI, i64 cSO, i64 cSN) {
  __shared__ float sA[2][64 * 32];   // 16 KB: 64 rows x 32 k, re+im
  __shared__ float sB[2][32 * 16];   //  4 KB: twiddle tile, re+im
  const int tid  = (int)threadIdx.x;
  const int wave = tid >> 5;
  const int lane = tid & 31;
  const int m0 = (int)blockIdx.x * 64;
  const int n0 = (int)blockIdx.y * 16;
  const bool hiHalf = (lane & 16) != 0;
  const bool cplxA = (Aim != nullptr);
  const int nComp = cplxA ? 2 : 1;
  const unsigned ldsA0 = (unsigned)(uintptr_t)(&sA[0][0]);

  v8f accP = {}; v8f accN = {}; v8f accI = {};

  for (int k0 = 0; k0 < K; k0 += 32) {
    // ---- stage B (twiddle) tile, coalesced ----
    for (int u = 0; u < 4; ++u) {
      int item = u * 128 + tid;                 // 0..511 == kk*16+nn
      int kk = item >> 4, nn = item & 15;
      int gk = k0 + kk, gn = n0 + nn;
      float tr = 0.f, ti = 0.f;
      if (gk < K && gn < N) { tr = Bre[(i64)gk * N + gn]; ti = Bim[(i64)gk * N + gn]; }
      sB[0][item] = tr;
      sB[1][item] = ti;
    }
    // ---- stage A tile (64 rows x 32 k) ----
    if (aSK == 1) {
      // contiguous k: async DMA, one B128 (4 floats) per op. K is a multiple of
      // 4 so chunks never straddle the edge; OOB chunks are zero-filled via DS.
      for (int comp = 0; comp < nComp; ++comp) {
        const float* src = comp ? Aim : Are;
        for (int u = 0; u < 4; ++u) {
          int item = u * 128 + tid;             // 0..511
          int row = item >> 3, chunk = item & 7;
          int gm = m0 + row;
          int gk = k0 + chunk * 4;
          if (gm < M && gk + 4 <= K) {
            i64 base = (i64)(gm / aInner) * aSO + (i64)(gm % aInner) * aSI;
            unsigned lds = ldsA0 + (unsigned)((comp * 2048 + row * 32 + chunk * 4) * 4);
            unsigned long long gaddr = (unsigned long long)(uintptr_t)(src + base + gk);
            asm volatile("global_load_async_to_lds_b128 %0, %1, off"
                         :: "v"(lds), "v"(gaddr) : "memory");
          } else {
            int o = row * 32 + chunk * 4;
            sA[comp][o + 0] = 0.f; sA[comp][o + 1] = 0.f;
            sA[comp][o + 2] = 0.f; sA[comp][o + 3] = 0.f;
          }
        }
      }
      asm volatile("s_wait_asynccnt 0x0" ::: "memory");
    } else {
      // strided k ("column DFT"): row stride aSI==1 -> consecutive threads walk
      // consecutive rows for coalescing.
      for (int comp = 0; comp < nComp; ++comp) {
        const float* src = comp ? Aim : Are;
        for (int u = 0; u < 16; ++u) {
          int item = u * 128 + tid;             // 0..2047
          int col = item >> 6, row = item & 63;
          int gm = m0 + row;
          int gk = k0 + col;
          float v = 0.f;
          if (gm < M && gk < K) {
            i64 base = (i64)(gm / aInner) * aSO + (i64)(gm % aInner) * aSI;
            v = src[base + (i64)gk * aSK];
            if (gk + 32 < K) __builtin_prefetch(src + base + (i64)(gk + 32) * aSK, 0, 1);
          }
          sA[comp][row * 32 + col] = v;
        }
      }
    }
    __syncthreads();

    // ---- fragment gather from LDS + WMMA ----
    v16h ar, ai, br, bi;
    const int khalf = hiHalf ? 8 : 0;
    const int rowl = wave * 16 + (lane & 15);
#pragma unroll
    for (int i = 0; i < 16; ++i) {
      int kk = (i < 8) ? (khalf + i) : (16 + khalf + (i - 8));
      ar[i] = (_Float16)sA[0][rowl * 32 + kk];
      ai[i] = (_Float16)sA[1][rowl * 32 + kk];
      br[i] = (_Float16)sB[0][kk * 16 + (lane & 15)];
      bi[i] = (_Float16)sB[1][kk * 16 + (lane & 15)];
    }
    // Cre = sum Ar*Br - sum Ai*Bi ; Cim = sum Ar*Bi + Ai*Br
    accP = __builtin_amdgcn_wmma_f32_16x16x32_f16(false, ar, false, br, (short)0, accP, false, false);
    accI = __builtin_amdgcn_wmma_f32_16x16x32_f16(false, ar, false, bi, (short)0, accI, false, false);
    if (cplxA) {
      accN = __builtin_amdgcn_wmma_f32_16x16x32_f16(false, ai, false, bi, (short)0, accN, false, false);
      accI = __builtin_amdgcn_wmma_f32_16x16x32_f16(false, ai, false, br, (short)0, accI, false, false);
    }
    __syncthreads();
  }

  const int nc = n0 + (lane & 15);
  if (nc < N) {
#pragma unroll
    for (int r = 0; r < 8; ++r) {
      int row = m0 + wave * 16 + r + (hiHalf ? 8 : 0);
      if (row < M) {
        i64 q = row / cInner, rr = row % cInner;
        i64 ad = q * cSO + rr * cSI + (i64)nc * cSN;
        Cre[ad] = accP[r] - accN[r];
        if (Cim) Cim[ad] = accI[r];
      }
    }
  }
}

// ---------------------------------------------------------------------------
// Spectral plumbing (ragged per-l; VALU kernels with closed-form offsets)
// ---------------------------------------------------------------------------
// Fx[l][m][b][c] = sum_j X[b,c,j,mi] * bank[l][j][m]
__global__ void sphcnn_s2_analysis(float* Fxre, float* Fxim, const float* Xre, const float* Xim,
                                   const float* bank, int nl, int Nf, int J, int B, int C, i64 total) {
  i64 idx = gidx();
  if (idx >= total) return;
  i64 row = idx / (B * C);
  int bc = (int)(idx % (B * C));
  int b = bc / C, c = bc % C;
  int l = 0;
  while (cum_lin(l + 1) <= row) ++l;
  int m = (int)(row - cum_lin(l)) - l;
  int d = 2 * l + 1;
  int mi = ((m % Nf) + Nf) % Nf;
  i64 boff = (i64)J * cum_lin(l);
  float sre = 0.f, sim = 0.f;
  for (int j = 0; j < J; ++j) {
    float w = bank[boff + (i64)j * d + (m + l)];
    i64 xi = (((i64)b * C + c) * J + j) * Nf + mi;
    sre += w * Xre[xi];
    sim += w * Xim[xi];
  }
  Fxre[idx] = sre; Fxim[idx] = sim;
}

// Fx[l][m][n][b][c] = sum_j X[b,c,j,mi,ni] * bank[l][j][m][n]
__global__ void sphcnn_so3_analysis(float* Fxre, float* Fxim, const float* Xre, const float* Xim,
                                    const float* bank, int nl, int Nf, int J, int B, int C, i64 total) {
  i64 idx = gidx();
  if (idx >= total) return;
  i64 row = idx / (B * C);
  int bc = (int)(idx % (B * C));
  int b = bc / C, c = bc % C;
  int l = 0;
  while (cum_sq(l + 1) <= row) ++l;
  i64 rem = row - cum_sq(l);
  int d = 2 * l + 1;
  int m = (int)(rem / d) - l, n = (int)(rem % d) - l;
  int mi = ((m % Nf) + Nf) % Nf, ni = ((n % Nf) + Nf) % Nf;
  i64 boff = (i64)J * cum_sq(l);
  float sre = 0.f, sim = 0.f;
  for (int j = 0; j < J; ++j) {
    float w = bank[boff + ((i64)j * d + (m + l)) * d + (n + l)];
    i64 xi = ((((i64)b * C + c) * J + j) * Nf + mi) * Nf + ni;
    sre += w * Xre[xi];
    sim += w * Xim[xi];
  }
  Fxre[idx] = sre; Fxim[idx] = sim;
}

// Fz[l][m][n][b][o] = sum_i Fx[l][m][b][i] * conj(Fy[l][n][i][o])
__global__ void sphcnn_s2_mix(float* Fzre, float* Fzim, const float* Fxre, const float* Fxim,
                              const float* Fyre, const float* Fyim, int nl, int B, int Ci, int Co, i64 total) {
  i64 idx = gidx();
  if (idx >= total) return;
  i64 row = idx / (B * Co);
  int bo = (int)(idx % (B * Co));
  int b = bo / Co, o = bo % Co;
  int l = 0;
  while (cum_sq(l + 1) <= row) ++l;
  i64 rem = row - cum_sq(l);
  int d = 2 * l + 1;
  int m = (int)(rem / d), n = (int)(rem % d); // 0..2l index directly
  i64 fxb = cum_lin(l) * (i64)B * Ci + ((i64)m * B + b) * Ci;
  i64 fyb = cum_lin(l) * (i64)Ci * Co + (i64)n * Ci * Co;
  float sre = 0.f, sim = 0.f;
  for (int i = 0; i < Ci; ++i) {
    float xr = Fxre[fxb + i], xi = Fxim[fxb + i];
    float yr = Fyre[fyb + (i64)i * Co + o], yi = Fyim[fyb + (i64)i * Co + o];
    sre += xr * yr + xi * yi;      // x * conj(y)
    sim += xi * yr - xr * yi;
  }
  Fzre[idx] = sre; Fzim[idx] = sim;
}

// Fz[l][m][k][b][o] = sum_{n,i} Fx[l][m][n][b][i] * conj(Fy[l][k][n][i][o])
__global__ void sphcnn_so3_mix(float* Fzre, float* Fzim, const float* Fxre, const float* Fxim,
                               const float* Fyre, const float* Fyim, int nl, int B, int Ci, int Co, i64 total) {
  i64 idx = gidx();
  if (idx >= total) return;
  i64 row = idx / (B * Co);
  int bo = (int)(idx % (B * Co));
  int b = bo / Co, o = bo % Co;
  int l = 0;
  while (cum_sq(l + 1) <= row) ++l;
  i64 rem = row - cum_sq(l);
  int d = 2 * l + 1;
  int m = (int)(rem / d), k = (int)(rem % d);
  i64 fx0 = cum_sq(l) * (i64)B * Ci;
  i64 fy0 = cum_sq(l) * (i64)Ci * Co;
  float sre = 0.f, sim = 0.f;
  for (int n = 0; n < d; ++n) {
    i64 fxb = fx0 + (((i64)m * d + n) * B + b) * Ci;
    i64 fyb = fy0 + ((i64)k * d + n) * Ci * Co;
    for (int i = 0; i < Ci; ++i) {
      float xr = Fxre[fxb + i], xi = Fxim[fxb + i];
      float yr = Fyre[fyb + (i64)i * Co + o], yi = Fyim[fyb + (i64)i * Co + o];
      sre += xr * yr + xi * yi;
      sim += xi * yr - xr * yi;
    }
  }
  Fzre[idx] = sre; Fzim[idx] = sim;
}

// X[b,c,j,mi,ni] = sum_{l>=max(|m|,|n|)} bankS[l][j][m][n] * Fz[l][m][n][b][c]
__global__ void sphcnn_scatter(float* Xre, float* Xim, const float* bankS,
                               const float* Fzre, const float* Fzim,
                               int nl, int Nf, int J, int B, int C, i64 total) {
  i64 idx = gidx();
  if (idx >= total) return;
  i64 t = idx;
  int ni = (int)(t % Nf); t /= Nf;
  int mi = (int)(t % Nf); t /= Nf;
  int j  = (int)(t % J);  t /= J;
  int c  = (int)(t % C);
  int b  = (int)(t / C);
  int m = (2 * mi <= Nf) ? mi : mi - Nf;
  int n = (2 * ni <= Nf) ? ni : ni - Nf;
  int am = m < 0 ? -m : m, an = n < 0 ? -n : n;
  int lmin = am > an ? am : an;
  float sre = 0.f, sim = 0.f;
  for (int l = lmin; l < nl; ++l) {
    int d = 2 * l + 1;
    float w = bankS[(i64)J * cum_sq(l) + ((i64)j * d + (m + l)) * d + (n + l)];
    i64 fo = cum_sq(l) * (i64)B * C + ((((i64)(m + l) * d + (n + l)) * B + b) * C + c);
    sre += w * Fzre[fo];
    sim += w * Fzim[fo];
  }
  Xre[idx] = sre;
  Xim[idx] = sim;
}

// ---------------------------------------------------------------------------
// PReLU + BatchNorm + final quadrature
// ---------------------------------------------------------------------------
__global__ void sphcnn_zero(float* p, i64 n) {
  i64 idx = gidx();
  if (idx < n) p[idx] = 0.f;
}

__global__ void sphcnn_prelu_stats(float* h, const float* a, float* stats,
                                   i64 spatial, int C, i64 total) {
  __shared__ float ls[128];
  int tid = (int)threadIdx.x;
  ls[tid] = 0.f;
  __syncthreads();
  i64 idx = (i64)blockIdx.x * blockDim.x + tid;
  if (idx < total) {
    float x = h[idx];
    float al = a[0];
    float y = x >= 0.f ? x : al * x;
    h[idx] = y;
    int c = (int)((idx / spatial) % C);
    atomicAdd(&ls[c], y);
    atomicAdd(&ls[64 + c], y * y);
  }
  __syncthreads();
  if (tid < C) atomicAdd(&stats[tid], ls[tid]);
  if (tid >= 64 && tid < 64 + C) atomicAdd(&stats[C + (tid - 64)], ls[tid]);
}

__global__ void sphcnn_bn_apply(float* h, const float* stats, const float* g, const float* bb,
                                i64 spatial, int C, i64 total, float invCnt) {
  i64 idx = gidx();
  if (idx >= total) return;
  int c = (int)((idx / spatial) % C);
  float mu = stats[c] * invCnt;
  float var = stats[C + c] * invCnt - mu * mu;
  h[idx] = g[c] * (h[idx] - mu) * rsqrtf(var + 1e-5f) + bb[c];
}

__global__ void sphcnn_final(const float* h, const float* w, float* out, int G, i64 nout) {
  i64 idx = gidx();
  if (idx >= nout) return;
  float s = 0.f;
  for (int g = 0; g < G; ++g) s += h[idx * G + g] * w[g];
  out[idx] = s;
}

// ---------------------------------------------------------------------------
// Driver
// ---------------------------------------------------------------------------
static inline dim3 grd(i64 n) { return dim3((unsigned)((n + 255) / 256)); }

static void gemm_launch(hipStream_t s,
                        const float* Are, const float* Aim,
                        const float* Bre, const float* Bim,
                        float* Cre, float* Cim,
                        int M, int N, int K,
                        int aInner, i64 aSI, i64 aSO, i64 aSK,
                        int cInner, i64 cSI, i64 cSO, i64 cSN) {
  dim3 g((M + 63) / 64, (N + 15) / 16), b(128);
  sphcnn_gemm_cplx_wmma<<<g, b, 0, s>>>(Are, Aim, Bre, Bim, Cre, Cim, M, N, K,
                                        aInner, aSI, aSO, aSK, cInner, cSI, cSO, cSN);
}

extern "C" void kernel_launch(void* const* d_in, const int* in_sizes, int n_in,
                              void* d_out, int out_size, void* d_ws, size_t ws_size,
                              hipStream_t stream) {
  (void)in_sizes; (void)n_in; (void)out_size; (void)ws_size;
  const float* x1 = (const float*)d_in[0];
  const float* k1 = (const float*)d_in[1];
  const float* k2 = (const float*)d_in[2];
  const float* k3 = (const float*)d_in[3];
  const float* k4 = (const float*)d_in[4];
  const float* a1 = (const float*)d_in[5];
  const float* a2 = (const float*)d_in[6];
  const float* a3 = (const float*)d_in[7];
  const float* a4 = (const float*)d_in[8];
  const float* g1 = (const float*)d_in[9];  const float* bb1 = (const float*)d_in[10];
  const float* g2 = (const float*)d_in[11]; const float* bb2 = (const float*)d_in[12];
  const float* g3 = (const float*)d_in[13]; const float* bb3 = (const float*)d_in[14];
  const float* g4 = (const float*)d_in[15]; const float* bb4 = (const float*)d_in[16];
  float* out = (float*)d_out;

  // workspace carving
  char* p = (char*)d_ws;
  double* lf = (double*)p; p += 128 * sizeof(double);
  auto F = [&](i64 n) { float* r = (float*)p; p += n * sizeof(float); return r; };

  float* qw30 = F(60); float* qw20 = F(40); float* qw10 = F(20);
  float* F60r = F(3600); float* F60i = F(3600); float* I60r = F(3600); float* I60i = F(3600);
  float* F40r = F(1600); float* F40i = F(1600); float* I40r = F(1600); float* I40i = F(1600);
  float* F20r = F(400);  float* F20i = F(400);  float* I20r = F(400);  float* I20i = F(400);

  const i64 SQ30 = 35990, SQ20 = 10660, SQ10 = 1330; // cum_sq(30/20/10)
  float* bkS2A = F(60 * 900);     // s2 analysis, b_in=30, nl=30
  float* bkSA  = F(60 * SQ30);    // synthesis b_inv=30, nl=30
  float* bkAB  = F(60 * SQ20);    // so3 analysis b_in=30, nl=20
  float* bkSB  = F(40 * SQ20);    // synthesis b_inv=20
  float* bkAC  = F(40 * SQ10);    // analysis b_in=20, nl=10
  float* bkSC  = F(20 * SQ10);    // synthesis b_inv=10
  float* bkAD  = F(20 * SQ10);    // analysis b_in=10, nl=10
  float* bkSD  = F(60 * SQ10);    // synthesis b_inv=30, nl=10

  float* FyAr = F(900 * 20);        float* FyAi = F(900 * 20);
  float* FyBr = F(SQ20 * 200);      float* FyBi = F(SQ20 * 200);
  float* FyCr = F(SQ10 * 1200);     float* FyCi = F(SQ10 * 1200);
  float* FyDr = F(SQ10 * 60);       float* FyDi = F(SQ10 * 60);

  float* FxAr = F(900 * 16);        float* FxAi = F(900 * 16);
  float* FzAr = F(SQ30 * 80);       float* FzAi = F(SQ30 * 80);
  float* FxBr = F(SQ20 * 80);       float* FxBi = F(SQ20 * 80);
  float* FzBr = F(SQ20 * 160);      float* FzBi = F(SQ20 * 160);
  float* FxCr = F(SQ10 * 160);      float* FxCi = F(SQ10 * 160);
  float* FzCr = F(SQ10 * 480);      float* FzCi = F(SQ10 * 480);
  float* FxDr = F(SQ10 * 480);      float* FxDi = F(SQ10 * 480);
  float* FzDr = F(SQ10 * 8);        float* FzDi = F(SQ10 * 8);

  float* XAr = F(57600); float* XAi = F(57600);
  float* stats = F(128);

  const i64 BIGT = 17280000; // 8*10*60^3
  float* P0r = F(BIGT); float* P0i = F(BIGT);
  float* P1r = F(BIGT); float* P1i = F(BIGT);
  float* hbuf = F(BIGT);

  // ---- tables -------------------------------------------------------------
  sphcnn_lfact<<<1, 1, 0, stream>>>(lf);
  sphcnn_quadw<<<grd(60), 256, 0, stream>>>(qw30, 30);
  sphcnn_quadw<<<grd(40), 256, 0, stream>>>(qw20, 20);
  sphcnn_quadw<<<grd(20), 256, 0, stream>>>(qw10, 10);
  sphcnn_dftmat<<<grd(3600), 256, 0, stream>>>(F60r, F60i, 60, -1);
  sphcnn_dftmat<<<grd(3600), 256, 0, stream>>>(I60r, I60i, 60, +1);
  sphcnn_dftmat<<<grd(1600), 256, 0, stream>>>(F40r, F40i, 40, -1);
  sphcnn_dftmat<<<grd(1600), 256, 0, stream>>>(I40r, I40i, 40, +1);
  sphcnn_dftmat<<<grd(400), 256, 0, stream>>>(F20r, F20i, 20, -1);
  sphcnn_dftmat<<<grd(400), 256, 0, stream>>>(I20r, I20i, 20, +1);

  sphcnn_s2_bank<<<grd(60 * 900), 256, 0, stream>>>(bkS2A, 60, 30, qw30, lf, 60 * 900);
  sphcnn_so3_bank<<<grd(60 * SQ30), 256, 0, stream>>>(bkSA, 60, 30, nullptr, 1, lf, 60 * SQ30);
  sphcnn_so3_bank<<<grd(60 * SQ20), 256, 0, stream>>>(bkAB, 60, 20, qw30, 0, lf, 60 * SQ20);
  sphcnn_so3_bank<<<grd(40 * SQ20), 256, 0, stream>>>(bkSB, 40, 20, nullptr, 1, lf, 40 * SQ20);
  sphcnn_so3_bank<<<grd(40 * SQ10), 256, 0, stream>>>(bkAC, 40, 10, qw20, 0, lf, 40 * SQ10);
  sphcnn_so3_bank<<<grd(20 * SQ10), 256, 0, stream>>>(bkSC, 20, 10, nullptr, 1, lf, 20 * SQ10);
  sphcnn_so3_bank<<<grd(20 * SQ10), 256, 0, stream>>>(bkAD, 20, 10, qw10, 0, lf, 20 * SQ10);
  sphcnn_so3_bank<<<grd(60 * SQ10), 256, 0, stream>>>(bkSD, 60, 10, nullptr, 1, lf, 60 * SQ10);

  sphcnn_spec_s2<<<grd(900 * 20), 256, 0, stream>>>(FyAr, FyAi, k1, 30, 2, 10,
      PI_D / 160.0, 6, 1.0f / sqrtf(12.f), lf, 900 * 20);
  sphcnn_spec_so3<<<grd(SQ20 * 200), 256, 0, stream>>>(FyBr, FyBi, k2, 20, 10, 20,
      PI_D / 16.0, 6, 6, 1.0f / sqrtf(360.f), lf, SQ20 * 200);
  sphcnn_spec_so3<<<grd(SQ10 * 1200), 256, 0, stream>>>(FyCr, FyCi, k3, 10, 20, 60,
      PI_D / 8.0, 6, 6, 1.0f / sqrtf(720.f), lf, SQ10 * 1200);
  sphcnn_spec_so3<<<grd(SQ10 * 60), 256, 0, stream>>>(FyDr, FyDi, k4, 10, 60, 1,
      PI_D / 4.0, 6, 6, 1.0f / sqrtf(2160.f), lf, SQ10 * 60);

  // ---- stage A: s2_conv(x1,k1,30,30) -> h1 (8,10,60,60,60) ----------------
  gemm_launch(stream, x1, nullptr, F60r, F60i, XAr, XAi, 960, 60, 60,
              BIGI, 60, 0, 1, BIGI, 60, 0, 1);
  sphcnn_s2_analysis<<<grd(900 * 16), 256, 0, stream>>>(FxAr, FxAi, XAr, XAi, bkS2A,
      30, 60, 60, 8, 2, 900 * 16);
  sphcnn_s2_mix<<<grd(SQ30 * 80), 256, 0, stream>>>(FzAr, FzAi, FxAr, FxAi, FyAr, FyAi,
      30, 8, 2, 10, SQ30 * 80);
  sphcnn_scatter<<<grd(BIGT), 256, 0, stream>>>(P0r, P0i, bkSA, FzAr, FzAi,
      30, 60, 60, 8, 10, BIGT);
  gemm_launch(stream, P0r, P0i, I60r, I60i, P1r, P1i, 288000, 60, 60,
              BIGI, 60, 0, 1, BIGI, 60, 0, 1);
  gemm_launch(stream, P1r, P1i, I60r, I60i, hbuf, nullptr, 288000, 60, 60,
              60, 1, 3600, 60, 60, 1, 3600, 60);
  sphcnn_zero<<<grd(20), 256, 0, stream>>>(stats, 20);
  sphcnn_prelu_stats<<<dim3((unsigned)((BIGT + 127) / 128)), 128, 0, stream>>>(hbuf, a1, stats, 216000, 10, BIGT);
  sphcnn_bn_apply<<<grd(BIGT), 256, 0, stream>>>(hbuf, stats, g1, bb1, 216000, 10, BIGT,
      1.0f / (8.f * 216000.f));

  // ---- stage B: so3_conv(h1,k2,20,20) -> h2 (8,20,40,40,40) ---------------
  gemm_launch(stream, hbuf, nullptr, F60r, F60i, P0r, P0i, 288000, 60, 60,
              BIGI, 60, 0, 1, BIGI, 60, 0, 1);
  gemm_launch(stream, P0r, P0i, F60r, F60i, P1r, P1i, 288000, 60, 60,
              60, 1, 3600, 60, 60, 1, 3600, 60);
  sphcnn_so3_analysis<<<grd(SQ20 * 80), 256, 0, stream>>>(FxBr, FxBi, P1r, P1i, bkAB,
      20, 60, 60, 8, 10, SQ20 * 80);
  sphcnn_so3_mix<<<grd(SQ20 * 160), 256, 0, stream>>>(FzBr, FzBi, FxBr, FxBi, FyBr, FyBi,
      20, 8, 10, 20, SQ20 * 160);
  const i64 T_B = 10240000; // 8*20*40^3
  sphcnn_scatter<<<grd(T_B), 256, 0, stream>>>(P0r, P0i, bkSB, FzBr, FzBi,
      20, 40, 40, 8, 20, T_B);
  gemm_launch(stream, P0r, P0i, I40r, I40i, P1r, P1i, 256000, 40, 40,
              BIGI, 40, 0, 1, BIGI, 40, 0, 1);
  gemm_launch(stream, P1r, P1i, I40r, I40i, hbuf, nullptr, 256000, 40, 40,
              40, 1, 1600, 40, 40, 1, 1600, 40);
  sphcnn_zero<<<grd(40), 256, 0, stream>>>(stats, 40);
  sphcnn_prelu_stats<<<dim3((unsigned)((T_B + 127) / 128)), 128, 0, stream>>>(hbuf, a2, stats, 64000, 20, T_B);
  sphcnn_bn_apply<<<grd(T_B), 256, 0, stream>>>(hbuf, stats, g2, bb2, 64000, 20, T_B,
      1.0f / (8.f * 64000.f));

  // ---- stage C: so3_conv(h2,k3,10,10) -> h3 (8,60,20,20,20) ---------------
  gemm_launch(stream, hbuf, nullptr, F40r, F40i, P0r, P0i, 256000, 40, 40,
              BIGI, 40, 0, 1, BIGI, 40, 0, 1);
  gemm_launch(stream, P0r, P0i, F40r, F40i, P1r, P1i, 256000, 40, 40,
              40, 1, 1600, 40, 40, 1, 1600, 40);
  sphcnn_so3_analysis<<<grd(SQ10 * 160), 256, 0, stream>>>(FxCr, FxCi, P1r, P1i, bkAC,
      10, 40, 40, 8, 20, SQ10 * 160);
  sphcnn_so3_mix<<<grd(SQ10 * 480), 256, 0, stream>>>(FzCr, FzCi, FxCr, FxCi, FyCr, FyCi,
      10, 8, 20, 60, SQ10 * 480);
  const i64 T_C = 3840000; // 8*60*20^3
  sphcnn_scatter<<<grd(T_C), 256, 0, stream>>>(P0r, P0i, bkSC, FzCr, FzCi,
      10, 20, 20, 8, 60, T_C);
  gemm_launch(stream, P0r, P0i, I20r, I20i, P1r, P1i, 192000, 20, 20,
              BIGI, 20, 0, 1, BIGI, 20, 0, 1);
  gemm_launch(stream, P1r, P1i, I20r, I20i, hbuf, nullptr, 192000, 20, 20,
              20, 1, 400, 20, 20, 1, 400, 20);
  sphcnn_zero<<<grd(120), 256, 0, stream>>>(stats, 120);
  sphcnn_prelu_stats<<<dim3((unsigned)((T_C + 127) / 128)), 128, 0, stream>>>(hbuf, a3, stats, 8000, 60, T_C);
  sphcnn_bn_apply<<<grd(T_C), 256, 0, stream>>>(hbuf, stats, g3, bb3, 8000, 60, T_C,
      1.0f / (8.f * 8000.f));

  // ---- stage D: so3_conv(h3,k4,10,30) -> h4 (8,1,60,60,60) ----------------
  gemm_launch(stream, hbuf, nullptr, F20r, F20i, P0r, P0i, 192000, 20, 20,
              BIGI, 20, 0, 1, BIGI, 20, 0, 1);
  gemm_launch(stream, P0r, P0i, F20r, F20i, P1r, P1i, 192000, 20, 20,
              20, 1, 400, 20, 20, 1, 400, 20);
  sphcnn_so3_analysis<<<grd(SQ10 * 480), 256, 0, stream>>>(FxDr, FxDi, P1r, P1i, bkAD,
      10, 20, 20, 8, 60, SQ10 * 480);
  sphcnn_so3_mix<<<grd(SQ10 * 8), 256, 0, stream>>>(FzDr, FzDi, FxDr, FxDi, FyDr, FyDi,
      10, 8, 60, 1, SQ10 * 8);
  const i64 T_D = 1728000; // 8*1*60^3
  sphcnn_scatter<<<grd(T_D), 256, 0, stream>>>(P0r, P0i, bkSD, FzDr, FzDi,
      10, 60, 60, 8, 1, T_D);
  gemm_launch(stream, P0r, P0i, I60r, I60i, P1r, P1i, 28800, 60, 60,
              BIGI, 60, 0, 1, BIGI, 60, 0, 1);
  gemm_launch(stream, P1r, P1i, I60r, I60i, hbuf, nullptr, 28800, 60, 60,
              60, 1, 3600, 60, 60, 1, 3600, 60);
  sphcnn_zero<<<grd(2), 256, 0, stream>>>(stats, 2);
  sphcnn_prelu_stats<<<dim3((unsigned)((T_D + 127) / 128)), 128, 0, stream>>>(hbuf, a4, stats, 216000, 1, T_D);
  sphcnn_bn_apply<<<grd(T_D), 256, 0, stream>>>(hbuf, stats, g4, bb4, 216000, 1, T_D,
      1.0f / (8.f * 216000.f));

  // ---- final quadrature over gamma ---------------------------------------
  sphcnn_final<<<grd(28800), 256, 0, stream>>>(hbuf, qw30, out, 60, 28800);
}
